// Downsample_PASA_group_softmax_1194000908422
// MI455X (gfx1250) — compile-verified, matrix operand order checked by reading, not run on hardware
//
#include <hip/hip_runtime.h>
#include <hip/hip_bf16.h>

typedef __attribute__((ext_vector_type(16))) _Float16 v16h;
typedef __attribute__((ext_vector_type(8)))  _Float16 v8h;
typedef __attribute__((ext_vector_type(8)))  float    v8f;

#define N_IMG   16
#define C_IN    64
#define H_IN    128
#define W_IN    128
#define H_OUT   64
#define W_OUT   64
#define GK2     18          // G*K*K softmax channels
#define NPAD    32          // N padded for two 16-wide WMMA tiles
#define KKDIM   576         // 9 taps * 64 channels (K = tap*64 + c)
#define KSTEPS  18          // 576 / 32
#define WAVES   4           // waves per workgroup
#define TILE_W  16          // output pixels per wave (WMMA M)

// ---- prep: swizzle conv weights into ws as bRow[n][K], K = tap*64 + c,
//      rows n = 18..31 zero-filled (removes all predication from hot loop) ----
__global__ void pasa_prep_weights(const float* __restrict__ cw,
                                  _Float16* __restrict__ bRow)
{
    int i = blockIdx.x * blockDim.x + threadIdx.x;
    if (i >= NPAD * KKDIM) return;
    int n = i / KKDIM, K = i % KKDIM;
    int tap = K >> 6, c = K & 63;
    _Float16 v = (_Float16)0.0f;
    if (n < GK2) v = (_Float16)cw[((n * C_IN + c) * 3 + tap / 3) * 3 + tap % 3];
    bRow[i] = v;
}

__global__ __launch_bounds__(WAVES * 32, 1)
void pasa_main(const float* __restrict__ x,
               const _Float16* __restrict__ bRow,
               const float* __restrict__ gamma,
               const float* __restrict__ beta,
               const float* __restrict__ mean,
               const float* __restrict__ var,
               float* __restrict__ out)
{
    // c innermost -> A fragments are contiguous 16B runs
    alignas(16) __shared__ _Float16 patchT[WAVES][3][34][C_IN];   // 4 x 13056 B
    __shared__ float sigLds[WAVES][TILE_W][GK2];                  // 4 x 1152 B
    __shared__ float sbLds[2 * GK2];                              // BN scale/bias

    const int tid    = threadIdx.x;
    const int lane   = tid & 31;
    const int wave   = tid >> 5;
    const int mRow   = lane & 15;
    const int laneHi = lane >> 4;

    if (tid < GK2) {
        float inv = gamma[tid] * rsqrtf(var[tid] + 1e-5f);
        sbLds[tid]       = inv;
        sbLds[GK2 + tid] = beta[tid] - mean[tid] * inv;
    }

    // ---- this wave's tile: image, output row, 16 output cols ----
    const int tile = blockIdx.x * WAVES + wave;    // 0 .. 4095
    const int wt  = tile & 3;
    const int h   = (tile >> 2) & 63;
    const int img = tile >> 8;
    const int w0  = wt * TILE_W;

    // ---- stage strided 3x34x64 patch (reflect: -1 -> 1) as f16, c innermost ----
    const float* xb = x + (long)img * C_IN * H_IN * W_IN;
    for (int i = lane; i < C_IN * 3 * 34; i += 32) {
        int col = i % 34;            // lanes walk cols -> coalesced global loads
        int t   = i / 34;
        int dy  = t % 3;
        int c   = t / 3;
        int ry = 2 * h + dy - 1;  if (ry < 0) ry = 1;
        int cx = 2 * w0 + col - 1; if (cx < 0) cx = 1; if (cx > 127) cx = 126;
        patchT[wave][dy][col][c] = (_Float16)xb[(c * H_IN + ry) * W_IN + cx];
    }
    __syncthreads();

    // ---- GEMM: sigma[16 px][18 ch] via v_wmma_f32_16x16x32_f16 ----
    v8f c0 = {};   // channels 0..15
    v8f c1 = {};   // channels 16..17 (+ zero pad)
    const _Float16* b0p = bRow + mRow * KKDIM;
    const _Float16* b1p = bRow + (16 + mRow) * KKDIM;
#pragma unroll
    for (int ks = 0; ks < KSTEPS; ++ks) {
        const int tap = ks >> 1;                 // compile-time after unroll
        const int dy  = tap / 3, dx = tap % 3;
        const int cb  = (ks & 1) * 32 + laneHi * 8;   // c base for this lane half
        // A: two 16B runs (K_local 0..7 and 16..23 for this lane half)
        const v8h* ap = (const v8h*)&patchT[wave][dy][2 * mRow + dx][cb];
        v16h a = __builtin_shufflevector(ap[0], ap[2],
                 0,1,2,3,4,5,6,7,8,9,10,11,12,13,14,15);
        // B: two 16B runs each from the pre-swizzled L2-resident weight rows
        const v8h* bp0 = (const v8h*)(b0p + (tap << 6) + cb);
        const v8h* bp1 = (const v8h*)(b1p + (tap << 6) + cb);
        v16h b0 = __builtin_shufflevector(bp0[0], bp0[2],
                 0,1,2,3,4,5,6,7,8,9,10,11,12,13,14,15);
        v16h b1 = __builtin_shufflevector(bp1[0], bp1[2],
                 0,1,2,3,4,5,6,7,8,9,10,11,12,13,14,15);
        c0 = __builtin_amdgcn_wmma_f32_16x16x32_f16(false, a, false, b0, (short)0, c0, false, false);
        c1 = __builtin_amdgcn_wmma_f32_16x16x32_f16(false, a, false, b1, (short)0, c1, false, false);
    }

    // ---- scatter C fragments (M = r + 8*laneHi, N = mRow) to LDS ----
#pragma unroll
    for (int r = 0; r < 8; ++r) {
        int m = r + (laneHi ? 8 : 0);
        sigLds[wave][m][mRow] = c0[r];
        if (mRow < 2) sigLds[wave][m][16 + mRow] = c1[r];
    }
    __syncthreads();

    // ---- per-pixel BN + softmax(18) ----
    const int p  = mRow;
    const int cg = laneHi;                 // channel half == group (C/G == 32)
    float prob[GK2];
    float mx = -1e30f;
#pragma unroll
    for (int i2 = 0; i2 < GK2; ++i2) {
        float v = sigLds[wave][p][i2] * sbLds[i2] + sbLds[GK2 + i2];
        prob[i2] = v;
        mx = fmaxf(mx, v);
    }
    float sum = 0.0f;
#pragma unroll
    for (int i2 = 0; i2 < GK2; ++i2) {
        float e = __expf(prob[i2] - mx);
        prob[i2] = e;
        sum += e;
    }
    const float rs = 1.0f / sum;

    // ---- softmax-weighted 3x3 aggregation, 32 channels per lane ----
    float acc[32];
#pragma unroll
    for (int cc = 0; cc < 32; ++cc) acc[cc] = 0.0f;
#pragma unroll
    for (int tap = 0; tap < 9; ++tap) {
        const int dy = tap / 3, dx = tap % 3;
        const float wgt = prob[cg * 9 + tap];
        const v8h* pp = (const v8h*)&patchT[wave][dy][2 * p + dx][cg * 32];
#pragma unroll
        for (int q = 0; q < 4; ++q) {
            v8h v = pp[q];
#pragma unroll
            for (int e = 0; e < 8; ++e) acc[q * 8 + e] += (float)v[e] * wgt;
        }
    }
    float* ob = out + (((long)img * C_IN) * H_OUT + h) * W_OUT + (w0 + p);
#pragma unroll
    for (int cc = 0; cc < 32; ++cc)
        ob[(long)(cg * 32 + cc) * H_OUT * W_OUT] = acc[cc] * rs;
}

extern "C" void kernel_launch(void* const* d_in, const int* in_sizes, int n_in,
                              void* d_out, int out_size, void* d_ws, size_t ws_size,
                              hipStream_t stream) {
    const float* x     = (const float*)d_in[0];
    const float* cw    = (const float*)d_in[1];
    const float* gamma = (const float*)d_in[2];
    const float* beta  = (const float*)d_in[3];
    const float* mean  = (const float*)d_in[4];
    const float* var   = (const float*)d_in[5];
    float* out = (float*)d_out;

    _Float16* bRow = (_Float16*)d_ws;              // 32*576 f16 = 36864 B

    pasa_prep_weights<<<(NPAD * KKDIM + 255) / 256, 256, 0, stream>>>(cw, bRow);

    const int tiles  = N_IMG * H_OUT * (W_OUT / TILE_W);  // 4096 wave tiles
    const int blocks = tiles / WAVES;                      // 1024 workgroups
    pasa_main<<<blocks, WAVES * 32, 0, stream>>>(x, bRow, gamma, beta, mean, var, out);
}
